// SPG_GCN_72310069395905
// MI455X (gfx1250) — matrix operand order repaired and checked by reading, not compiled, not run on hardware
//
#include <hip/hip_runtime.h>
#include <hip/hip_bf16.h>

#define DEV static __device__ __forceinline__

typedef __attribute__((ext_vector_type(16))) __bf16 v16bf;
typedef __attribute__((ext_vector_type(8)))  float  v8f;
typedef __attribute__((ext_vector_type(8)))  unsigned int v8u;
typedef __attribute__((ext_vector_type(4)))  unsigned int v4u;

namespace {
constexpr int   N_NODES = 50000;
constexpr int   N_EDGES = 250000;
constexpr int   IN_C    = 32;
constexpr int   HIDQ    = 16;
constexpr int   EDGE_DIM= 13;
constexpr int   N_CLS   = 13;
constexpr int   ITERS   = 10;
constexpr float EPSV    = 1e-5f;
constexpr int   HIST_W  = 192;   // (ITERS+1)*16 = 176, padded to 6 K-chunks of 32
}

DEV unsigned short f2bf(float f) {           // native v_cvt path
  __bf16 b = (__bf16)f;
  return __builtin_bit_cast(unsigned short, b);
}
DEV float bf2f(unsigned short h) { return __uint_as_float(((unsigned)h) << 16); }
DEV float sigmoidf(float x) { return 1.0f / (1.0f + __expf(-x)); }

DEV float rowsum16(float x) {                 // reduce across each 16-lane half
  x += __shfl_xor(x, 1, 32);
  x += __shfl_xor(x, 2, 32);
  x += __shfl_xor(x, 4, 32);
  x += __shfl_xor(x, 8, 32);
  return x;
}

// A-operand gather, 16-bit 16x32 layout (ISA 7.12.2):
// lane(0-15): M=lane, v0..3 -> K{0..7}, v4..7 -> K{16..23}; lane(16-31): +8 / +24.
// Each lane's 8 dwords are two contiguous 16B runs -> two b128 loads.
DEV v16bf load_A16(const unsigned short* base, int ldw, int kbase, int lane) {
  const int m = lane & 15, half = lane >> 4;
  const unsigned* row32 = (const unsigned*)(base + m * ldw + kbase);
  const v4u lo = *(const v4u*)(row32 + half * 4);        // K = half*8 + 0..7
  const v4u hi = *(const v4u*)(row32 + 8 + half * 4);    // K = 16 + half*8 + 0..7
  v8u u = __builtin_shufflevector(lo, hi, 0, 1, 2, 3, 4, 5, 6, 7);
  return __builtin_bit_cast(v16bf, u);
}

// B operand: packed so each lane reads its 16 bf16 contiguously (32B vector load)
DEV v16bf load_Bt(const unsigned short* tiles, int tileIdx, int lane) {
  const v8u* p = (const v8u*)(tiles + ((size_t)tileIdx * 32 + lane) * 16);
  return __builtin_bit_cast(v16bf, *p);
}

DEV v8f wmma_bf16(v16bf a, v16bf b, v8f c) {
  return __builtin_amdgcn_wmma_f32_16x16x32_bf16(false, a, false, b, (short)0, c, false, false);
}

// ---------------- utility kernels ----------------

__global__ void zero_f32(float* p, long n) {
  long i = (long)blockIdx.x * blockDim.x + threadIdx.x;
  if (i < n) p[i] = 0.f;
}

// per-column sum / sumsq over rows (C <= 32)
__global__ void __launch_bounds__(256) col_stats(const float* __restrict__ x, int R, int C,
                                                 float* __restrict__ sum, float* __restrict__ sumsq) {
  __shared__ float s1[256], s2[256];
  const int c = threadIdx.x & 31;
  const int g = threadIdx.x >> 5;             // 8 row-groups
  float a = 0.f, b = 0.f;
  if (c < C) {
    for (int r = blockIdx.x * 8 + g; r < R; r += gridDim.x * 8) {
      float v = x[(size_t)r * C + c];
      a += v; b += v * v;
    }
  }
  s1[threadIdx.x] = a; s2[threadIdx.x] = b;
  __syncthreads();
  if (g == 0 && c < C) {
    for (int i = 1; i < 8; ++i) { a += s1[i * 32 + c]; b += s2[i * 32 + c]; }
    atomicAdd(&sum[c], a); atomicAdd(&sumsq[c], b);
  }
}

// pack W[outN,inK] (row-major, B[k,n] = W[n,k]) into bf16 B tiles, zero-padded
__global__ void pack_B_kernel(const float* __restrict__ W, int outN, int inK,
                              unsigned short* __restrict__ dst, int ktiles) {
  const int t = blockIdx.x;                   // t = ntile*ktiles + ktile
  const int ntile = t / ktiles, ktile = t % ktiles;
  const int lane = threadIdx.x;
  const int nl = lane & 15, khalf = lane >> 4;
  const int n = ntile * 16 + nl;
#pragma unroll
  for (int j = 0; j < 16; ++j) {
    const int k = ktile * 32 + (j < 8 ? khalf * 8 + j : 16 + khalf * 8 + (j - 8));
    const float v = (n < outN && k < inK) ? W[(size_t)n * inK + k] : 0.f;
    dst[((size_t)t * 32 + lane) * 16 + j] = f2bf(v);
  }
}

__global__ void deg_kernel(const int* __restrict__ ei, float* __restrict__ deg) {
  int e = blockIdx.x * blockDim.x + threadIdx.x;
  if (e >= N_EDGES) return;
  atomicAdd(&deg[ei[N_EDGES + e]], 1.f);
}

// ---------------- fnet: per-edge weight matrices, 74 WMMAs / 16-edge tile ----------------
__global__ void __launch_bounds__(256) fnet_kernel(
    const float* __restrict__ eattr, const float* __restrict__ stats,
    const float* __restrict__ f_bn_g, const float* __restrict__ f_bn_b,
    const unsigned short* __restrict__ tiles,
    const float* __restrict__ f_b1, const float* __restrict__ f_b2, const float* __restrict__ f_b3,
    unsigned short* __restrict__ We, int etiles) {
  __shared__ alignas(64) unsigned short sA[8][16 * 32];
  __shared__ alignas(64) unsigned short sB[8][16 * 128];
  const int wave = threadIdx.x >> 5, lane = threadIdx.x & 31;
  const int tile = blockIdx.x * 8 + wave;
  if (tile >= etiles) return;                 // wave-uniform: EXEC stays all-ones for WMMA
  unsigned short* A = sA[wave];
  unsigned short* B = sB[wave];
  const int e0 = tile * 16;
  const float invE = 1.0f / (float)N_EDGES;
  const int col = lane & 15, rbase = (lane >> 4) * 8;

  // stage BN(edge_attr) -> A [16][32] (K padded)
  for (int idx = lane; idx < 16 * 32; idx += 32) {
    const int r = idx >> 5, k = idx & 31;
    float v = 0.f;
    if (k < EDGE_DIM) {
      const float m   = stats[96 + k] * invE;
      const float var = stats[112 + k] * invE - m * m;
      const float s   = f_bn_g[k] * rsqrtf(var + EPSV);
      v = s * (eattr[(size_t)(e0 + r) * EDGE_DIM + k] - m) + f_bn_b[k];
    }
    A[idx] = f2bf(v);
  }
  const unsigned short* Bf1 = tiles + (size_t)1  * 512;
  const unsigned short* Bf2 = tiles + (size_t)3  * 512;
  const unsigned short* Bf3 = tiles + (size_t)11 * 512;

  { // L1: 13->32
    v16bf a = load_A16(A, 32, 0, lane);
#pragma unroll
    for (int nt = 0; nt < 2; ++nt) {
      v8f c = {};
      c = wmma_bf16(a, load_Bt(Bf1, nt, lane), c);
#pragma unroll
      for (int v = 0; v < 8; ++v) {
        const float f = fmaxf(c[v] + f_b1[nt * 16 + col], 0.f);
        B[(rbase + v) * 128 + nt * 16 + col] = f2bf(f);
      }
    }
  }
  { // L2: 32->128 (input cols 0..31 of B already in regs before overwrite)
    v16bf a = load_A16(B, 128, 0, lane);
#pragma unroll
    for (int nt = 0; nt < 8; ++nt) {
      v8f c = {};
      c = wmma_bf16(a, load_Bt(Bf2, nt, lane), c);
#pragma unroll
      for (int v = 0; v < 8; ++v) {
        const float f = fmaxf(c[v] + f_b2[nt * 16 + col], 0.f);
        B[(rbase + v) * 128 + nt * 16 + col] = f2bf(f);
      }
    }
  }
  { // L3: 128->256, store W_e k-major bf16 [E][16][16] (L2-resident: 128MB)
    v16bf a0 = load_A16(B, 128, 0,  lane);
    v16bf a1 = load_A16(B, 128, 32, lane);
    v16bf a2 = load_A16(B, 128, 64, lane);
    v16bf a3 = load_A16(B, 128, 96, lane);
#pragma unroll
    for (int nt = 0; nt < 16; ++nt) {
      v8f c = {};
      c = wmma_bf16(a0, load_Bt(Bf3, nt * 4 + 0, lane), c);
      c = wmma_bf16(a1, load_Bt(Bf3, nt * 4 + 1, lane), c);
      c = wmma_bf16(a2, load_Bt(Bf3, nt * 4 + 2, lane), c);
      c = wmma_bf16(a3, load_Bt(Bf3, nt * 4 + 3, lane), c);
#pragma unroll
      for (int v = 0; v < 8; ++v) {
        const float f = c[v] + f_b3[nt * 16 + col];
        We[(size_t)(e0 + rbase + v) * 256 + nt * 16 + col] = f2bf(f);
      }
    }
  }
}

// ---------------- encoder ----------------
__global__ void __launch_bounds__(256) enc1_kernel(
    const float* __restrict__ x, const float* __restrict__ stats,
    const float* __restrict__ g1, const float* __restrict__ b1,
    const unsigned short* __restrict__ tiles, const float* __restrict__ enc_b,
    float* __restrict__ hpre, float* __restrict__ stats_out, int ntiles) {
  __shared__ alignas(64) unsigned short sA[8][16 * 32];
  const int wave = threadIdx.x >> 5, lane = threadIdx.x & 31;
  const int tile = blockIdx.x * 8 + wave;
  if (tile >= ntiles) return;
  unsigned short* A = sA[wave];
  const int n0 = tile * 16;
  const float invN = 1.0f / (float)N_NODES;
  for (int idx = lane; idx < 16 * 32; idx += 32) {
    const int r = idx >> 5, k = idx & 31;
    const float m   = stats[k] * invN;
    const float var = stats[32 + k] * invN - m * m;
    const float s   = g1[k] * rsqrtf(var + EPSV);
    A[idx] = f2bf(s * (x[(size_t)(n0 + r) * IN_C + k] - m) + b1[k]);
  }
  const int col = lane & 15, rbase = (lane >> 4) * 8;
  v16bf a = load_A16(A, 32, 0, lane);
  v8f c = {};
  c = wmma_bf16(a, load_Bt(tiles, 0, lane), c);   // enc_W tile 0, K=32 exact
  float s = 0.f, s2 = 0.f;
#pragma unroll
  for (int v = 0; v < 8; ++v) {
    const float f = c[v] + enc_b[col];
    hpre[(size_t)(n0 + rbase + v) * HIDQ + col] = f;
    s += f; s2 += f * f;
  }
  s  += __shfl_xor(s, 16, 32);
  s2 += __shfl_xor(s2, 16, 32);
  if (lane < 16) {                                 // BN2 batch-stats accumulation
    atomicAdd(&stats_out[64 + col], s);
    atomicAdd(&stats_out[80 + col], s2);
  }
}

__global__ void enc2_kernel(const float* __restrict__ hpre, const float* __restrict__ stats,
                            const float* __restrict__ g2, const float* __restrict__ b2,
                            float* __restrict__ h, unsigned short* __restrict__ hist) {
  int tid = blockIdx.x * 256 + threadIdx.x;
  if (tid >= N_NODES * HIDQ) return;
  const int n = tid >> 4, c = tid & 15;
  const float invN = 1.0f / (float)N_NODES;
  const float m   = stats[64 + c] * invN;
  const float var = stats[80 + c] * invN - m * m;
  float v = g2[c] * (hpre[tid] - m) * rsqrtf(var + EPSV) + b2[c];
  v = fmaxf(v, 0.f);
  h[tid] = v;
  hist[(size_t)n * HIST_W + c] = f2bf(v);
  hist[(size_t)n * HIST_W + 176 + c] = 0;          // zero classifier K-pad
}

// ---------------- message pass: per-edge matvec + scatter-mean (VALU; W_e from L2) ----------------
__global__ void __launch_bounds__(256) edge_msg_kernel(
    const int* __restrict__ ei, const float* __restrict__ h,
    const unsigned short* __restrict__ We, float* __restrict__ agg) {
  const int tid = blockIdx.x * blockDim.x + threadIdx.x;
  const int e = tid >> 4, o = tid & 15;
  if (e >= N_EDGES) return;
  __builtin_prefetch(We + (size_t)(e + 128) * 256, 0, 0);
  const int s = ei[e];
  const int d = ei[N_EDGES + e];
  const float4* h4 = (const float4*)(h + (size_t)s * HIDQ);
  const float4 h0 = h4[0], h1 = h4[1], h2 = h4[2], h3 = h4[3];
  const unsigned short* w = We + (size_t)e * 256 + o;
  float acc = 0.f;
  acc += h0.x * bf2f(w[0  * 16]); acc += h0.y * bf2f(w[1  * 16]);
  acc += h0.z * bf2f(w[2  * 16]); acc += h0.w * bf2f(w[3  * 16]);
  acc += h1.x * bf2f(w[4  * 16]); acc += h1.y * bf2f(w[5  * 16]);
  acc += h1.z * bf2f(w[6  * 16]); acc += h1.w * bf2f(w[7  * 16]);
  acc += h2.x * bf2f(w[8  * 16]); acc += h2.y * bf2f(w[9  * 16]);
  acc += h2.z * bf2f(w[10 * 16]); acc += h2.w * bf2f(w[11 * 16]);
  acc += h3.x * bf2f(w[12 * 16]); acc += h3.y * bf2f(w[13 * 16]);
  acc += h3.z * bf2f(w[14 * 16]); acc += h3.w * bf2f(w[15 * 16]);
  atomicAdd(&agg[(size_t)d * HIDQ + o], acc);
}

// ---------------- GRUCellEx: 7 WMMAs / 16-node tile, LN via shfl butterflies ----------------
__global__ void __launch_bounds__(256) gru_kernel(
    const float* __restrict__ agg, const float* __restrict__ deg,
    float* __restrict__ h, unsigned short* __restrict__ hist,
    const unsigned short* __restrict__ tiles,
    const float* __restrict__ gate_b, const float* __restrict__ lni_g, const float* __restrict__ lni_b,
    const float* __restrict__ lnh_g, const float* __restrict__ lnh_b,
    const float* __restrict__ b_ih, const float* __restrict__ b_hh,
    const float* __restrict__ ecc_bias, int ntiles, int t) {
  __shared__ alignas(64) unsigned short sA[8][16 * 32];
  const int wave = threadIdx.x >> 5, lane = threadIdx.x & 31;
  const int tile = blockIdx.x * 8 + wave;
  if (tile >= ntiles) return;
  unsigned short* A = sA[wave];
  const int col = lane & 15, rbase = (lane >> 4) * 8;
  const int n0 = tile * 16;
  const unsigned short* Bgate = tiles + (size_t)75 * 512;
  const unsigned short* Bih   = tiles + (size_t)76 * 512;
  const unsigned short* Bhh   = tiles + (size_t)79 * 512;

  float hD[8], inp[8];
#pragma unroll
  for (int v = 0; v < 8; ++v) {
    const int node = n0 + rbase + v;
    hD[v] = h[(size_t)node * HIDQ + col];
    float dg = deg[node]; dg = dg < 1.f ? 1.f : dg;
    inp[v] = agg[(size_t)node * HIDQ + col] / dg + ecc_bias[col];
  }
#pragma unroll
  for (int v = 0; v < 8; ++v) {                     // stage h (bf16, K-pad zero)
    A[(rbase + v) * 32 + col]      = f2bf(hD[v]);
    A[(rbase + v) * 32 + col + 16] = 0;
  }
  v16bf a = load_A16(A, 32, 0, lane);
  v8f g = {};
  g = wmma_bf16(a, load_Bt(Bgate, 0, lane), g);

  float gi_x[8], h_ln[8];
#pragma unroll
  for (int v = 0; v < 8; ++v) {
    const float gate = sigmoidf(g[v] + gate_b[col]);
    float tt  = gate * inp[v];
    float m   = rowsum16(tt) * (1.f / 16.f);
    float dd  = tt - m;
    float var = rowsum16(dd * dd) * (1.f / 16.f);
    gi_x[v] = lni_g[col] * dd * rsqrtf(var + EPSV) + lni_b[col];
    m   = rowsum16(hD[v]) * (1.f / 16.f);
    dd  = hD[v] - m;
    var = rowsum16(dd * dd) * (1.f / 16.f);
    h_ln[v] = lnh_g[col] * dd * rsqrtf(var + EPSV) + lnh_b[col];
  }
#pragma unroll
  for (int v = 0; v < 8; ++v) A[(rbase + v) * 32 + col] = f2bf(gi_x[v]);
  a = load_A16(A, 32, 0, lane);
  v8f ir = {}, iz = {}, in_ = {};
  ir  = wmma_bf16(a, load_Bt(Bih, 0, lane), ir);
  iz  = wmma_bf16(a, load_Bt(Bih, 1, lane), iz);
  in_ = wmma_bf16(a, load_Bt(Bih, 2, lane), in_);
#pragma unroll
  for (int v = 0; v < 8; ++v) A[(rbase + v) * 32 + col] = f2bf(h_ln[v]);
  a = load_A16(A, 32, 0, lane);
  v8f hr = {}, hz = {}, hn = {};
  hr = wmma_bf16(a, load_Bt(Bhh, 0, lane), hr);
  hz = wmma_bf16(a, load_Bt(Bhh, 1, lane), hz);
  hn = wmma_bf16(a, load_Bt(Bhh, 2, lane), hn);
#pragma unroll
  for (int v = 0; v < 8; ++v) {
    const float r  = sigmoidf(ir[v] + b_ih[col]      + hr[v] + b_hh[col]);
    const float z  = sigmoidf(iz[v] + b_ih[16 + col] + hz[v] + b_hh[16 + col]);
    const float nn = tanhf(in_[v] + b_ih[32 + col] + r * (hn[v] + b_hh[32 + col]));
    const float hnew = (1.f - z) * nn + z * h_ln[v];
    const int node = n0 + rbase + v;
    h[(size_t)node * HIDQ + col] = hnew;
    hist[(size_t)node * HIST_W + (t + 1) * 16 + col] = f2bf(hnew);
  }
}

// ---------------- classifier: 6 WMMAs / 16-node tile ----------------
__global__ void __launch_bounds__(256) cls_kernel(
    const unsigned short* __restrict__ hist, const unsigned short* __restrict__ tiles,
    const float* __restrict__ cls_b, float* __restrict__ out, int ntiles) {
  const int wave = threadIdx.x >> 5, lane = threadIdx.x & 31;
  const int tile = blockIdx.x * 8 + wave;
  if (tile >= ntiles) return;
  const int col = lane & 15, rbase = (lane >> 4) * 8;
  const int n0 = tile * 16;
  const unsigned short* Bcls = tiles + (size_t)82 * 512;
  v8f c = {};
#pragma unroll
  for (int kc = 0; kc < 6; ++kc) {
    v16bf a = load_A16(hist + (size_t)n0 * HIST_W, HIST_W, kc * 32, lane);
    c = wmma_bf16(a, load_Bt(Bcls, kc, lane), c);
  }
  if (col < N_CLS) {
#pragma unroll
    for (int v = 0; v < 8; ++v)
      out[(size_t)(n0 + rbase + v) * N_CLS + col] = c[v] + cls_b[col];
  }
}

// ---------------- host ----------------
extern "C" void kernel_launch(void* const* d_in, const int* in_sizes, int n_in,
                              void* d_out, int out_size, void* d_ws, size_t ws_size,
                              hipStream_t stream) {
  (void)in_sizes; (void)n_in; (void)out_size; (void)ws_size;
  const float* x         = (const float*)d_in[0];
  const int*   ei        = (const int*)  d_in[1];
  const float* eattr     = (const float*)d_in[2];
  const float* enc_bn1_g = (const float*)d_in[3];
  const float* enc_bn1_b = (const float*)d_in[4];
  const float* enc_W     = (const float*)d_in[5];
  const float* enc_b     = (const float*)d_in[6];
  const float* enc_bn2_g = (const float*)d_in[7];
  const float* enc_bn2_b = (const float*)d_in[8];
  const float* f_bn_g    = (const float*)d_in[9];
  const float* f_bn_b    = (const float*)d_in[10];
  const float* f_W1      = (const float*)d_in[11];
  const float* f_b1      = (const float*)d_in[12];
  const float* f_W2      = (const float*)d_in[13];
  const float* f_b2      = (const float*)d_in[14];
  const float* f_W3      = (const float*)d_in[15];
  const float* f_b3      = (const float*)d_in[16];
  const float* ecc_bias  = (const float*)d_in[17];
  const float* gate_W    = (const float*)d_in[18];
  const float* gate_b    = (const float*)d_in[19];
  const float* lni_g     = (const float*)d_in[20];
  const float* lni_b     = (const float*)d_in[21];
  const float* lnh_g     = (const float*)d_in[22];
  const float* lnh_b     = (const float*)d_in[23];
  const float* W_ih      = (const float*)d_in[24];
  const float* b_ih      = (const float*)d_in[25];
  const float* W_hh      = (const float*)d_in[26];
  const float* b_hh      = (const float*)d_in[27];
  const float* cls_W     = (const float*)d_in[28];
  const float* cls_b     = (const float*)d_in[29];

  char* ws = (char*)d_ws;
  size_t off = 0;
  unsigned short* We   = (unsigned short*)(ws + off); off += (size_t)N_EDGES * 256 * 2;   // 128 MB, L2-resident
  unsigned short* hist = (unsigned short*)(ws + off); off += (size_t)N_NODES * HIST_W * 2;
  float* h    = (float*)(ws + off); off += (size_t)N_NODES * HIDQ * 4;
  float* agg  = (float*)(ws + off); off += (size_t)N_NODES * HIDQ * 4;
  float* hpre = (float*)(ws + off); off += (size_t)N_NODES * HIDQ * 4;
  float* deg  = (float*)(ws + off); off += (size_t)N_NODES * 4;
  float* stats= (float*)(ws + off); off += 512;                                           // 128 f32 accumulators
  unsigned short* tiles = (unsigned short*)(ws + off); off += (size_t)88 * 1024;          // packed B tiles

  const int ETILES = N_EDGES / 16, NTILES = N_NODES / 16;

  // zero deg + stats (contiguous), then batch stats
  zero_f32<<<(N_NODES + 128 + 255) / 256, 256, 0, stream>>>(deg, N_NODES + 128);
  col_stats<<<512, 256, 0, stream>>>(x,     N_NODES, IN_C,     stats + 0,  stats + 32);
  col_stats<<<512, 256, 0, stream>>>(eattr, N_EDGES, EDGE_DIM, stats + 96, stats + 112);

  // pack all weights into bf16 B-tile layout
  pack_B_kernel<<<1,  32, 0, stream>>>(enc_W,  16,  32,  tiles + (size_t)0  * 512, 1);
  pack_B_kernel<<<2,  32, 0, stream>>>(f_W1,   32,  13,  tiles + (size_t)1  * 512, 1);
  pack_B_kernel<<<8,  32, 0, stream>>>(f_W2,  128,  32,  tiles + (size_t)3  * 512, 1);
  pack_B_kernel<<<64, 32, 0, stream>>>(f_W3,  256, 128,  tiles + (size_t)11 * 512, 4);
  pack_B_kernel<<<1,  32, 0, stream>>>(gate_W, 16,  16,  tiles + (size_t)75 * 512, 1);
  pack_B_kernel<<<3,  32, 0, stream>>>(W_ih,   48,  16,  tiles + (size_t)76 * 512, 1);
  pack_B_kernel<<<3,  32, 0, stream>>>(W_hh,   48,  16,  tiles + (size_t)79 * 512, 1);
  pack_B_kernel<<<6,  32, 0, stream>>>(cls_W,  13, 176,  tiles + (size_t)82 * 512, 6);

  deg_kernel<<<(N_EDGES + 255) / 256, 256, 0, stream>>>(ei, deg);

  // per-edge weight matrices (WMMA-heavy, done once)
  fnet_kernel<<<(ETILES + 7) / 8, 256, 0, stream>>>(eattr, stats, f_bn_g, f_bn_b, tiles,
                                                    f_b1, f_b2, f_b3, We, ETILES);
  // encoder
  enc1_kernel<<<(NTILES + 7) / 8, 256, 0, stream>>>(x, stats, enc_bn1_g, enc_bn1_b, tiles,
                                                    enc_b, hpre, stats, NTILES);
  enc2_kernel<<<(N_NODES * HIDQ) / 256, 256, 0, stream>>>(hpre, stats, enc_bn2_g, enc_bn2_b, h, hist);

  // message-passing iterations
  for (int t = 0; t < ITERS; ++t) {
    zero_f32<<<(N_NODES * HIDQ + 255) / 256, 256, 0, stream>>>(agg, (long)N_NODES * HIDQ);
    edge_msg_kernel<<<(N_EDGES * 16) / 256, 256, 0, stream>>>(ei, h, We, agg);
    gru_kernel<<<(NTILES + 7) / 8, 256, 0, stream>>>(agg, deg, h, hist, tiles, gate_b,
                                                     lni_g, lni_b, lnh_g, lnh_b,
                                                     b_ih, b_hh, ecc_bias, NTILES, t);
  }

  cls_kernel<<<(NTILES + 7) / 8, 256, 0, stream>>>(hist, tiles, cls_b, (float*)d_out, NTILES);
}
